// BertMaskRCNN1D_15874199126296
// MI455X (gfx1250) — compile-verified, matrix operand order checked by reading, not run on hardware
//
#include <hip/hip_runtime.h>
#include <hip/hip_bf16.h>
#include <math.h>

#define D_    384
#define B_    32
#define T_    512
#define A_    7
#define TOPK_ 64
#define KEEP_ 16
#define P_    16
#define NCAND (T_*A_)        // 3584 anchors per batch
#define NTN   (D_/16)        // 24 N-tiles for D=384
#define KS_D  (D_/32)        // 12 K-steps for K=384
#define KS_FPN (4*KS_D)      // 48 K-steps for the fused gated FPN (K=1536)
#define KS_CONV (3*KS_D)     // 36 K-steps for im2col 3-tap conv (K=1152)

// d_out layout (floats, concatenated in reference return order)
#define OFF_PROP   0                       // (32,16,2)   = 1024
#define OFF_SCORES 1024                    // (32,16)     = 512
#define OFF_CLS    1536                    // (32,16,8)   = 4096
#define OFF_BOX    5632                    // (32,16,2)   = 1024
#define OFF_MASK   6656                    // (32,16,16,1)= 8192
#define OFF_LEN    14848                   // (32,)       = 32

typedef __attribute__((ext_vector_type(16))) __bf16 v16bf;
typedef __attribute__((ext_vector_type(8)))  float  v8f;

// ---------------------------------------------------------------------------
// WMMA helpers (CDNA5 wave32, 16x16x32 bf16 -> f32)
// ---------------------------------------------------------------------------
__device__ __forceinline__ v8f wmma_bf16(v16bf a, v16bf b, v8f c) {
  return __builtin_amdgcn_wmma_f32_16x16x32_bf16(
      false, a, false, b, (short)0, c, false, false);
}

// A fragment: rows row0..row0+15, cols k0..k0+31 of row-major fp32 [.,ld]
__device__ __forceinline__ v16bf load_a_f32(const float* __restrict__ src, int ld,
                                            int row0, int k0) {
  int lane = threadIdx.x & 31;
  int m = lane & 15;
  int half = lane >> 4;
  const float* base = src + (size_t)(row0 + m) * ld + k0 + half * 8;
  v16bf a;
#pragma unroll
  for (int p = 0; p < 8; ++p) {
    int k = (p < 4) ? 2 * p : 2 * p + 8;
    a[2 * p]     = (__bf16)base[k];
    a[2 * p + 1] = (__bf16)base[k + 1];
  }
  return a;
}

// Prepacked B fragment: tile index = tn*ksteps + ks; per-lane 16 contiguous bf16.
__device__ __forceinline__ v16bf load_b_packed(const __bf16* __restrict__ Bp, int tile) {
  int lane = threadIdx.x & 31;
  return *(const v16bf*)(Bp + ((size_t)tile * 32 + lane) * 16);
}

// D store: f32 16x16 tile; lanes 0-15: M=v, lanes 16-31: M=v+8
__device__ __forceinline__ void store_d(float* __restrict__ dst, int ld,
                                        int row0, int n0, v8f c, bool relu) {
  int lane = threadIdx.x & 31;
  int n = n0 + (lane & 15);
  int mb = (lane >> 4) * 8;
#pragma unroll
  for (int v = 0; v < 8; ++v) {
    float x = c[v];
    if (relu) x = fmaxf(x, 0.f);
    dst[(size_t)(row0 + mb + v) * ld + n] = x;
  }
}

// ---------------------------------------------------------------------------
// Prep 0a: pack gated FPN weights (gate softmax folded into W and bias)
// dst layout: [(tn*48 + ks)*32 + lane][16] bf16, ks = mat*12 + klocal
// ---------------------------------------------------------------------------
__global__ void fpn_pack_kernel(const float* __restrict__ fpnW, const float* __restrict__ fpnB,
                                const float* __restrict__ gate,
                                __bf16* __restrict__ Bp, float* __restrict__ cbias) {
  float g0 = gate[0], g1 = gate[1], g2 = gate[2], g3 = gate[3];
  float mx = fmaxf(fmaxf(g0, g1), fmaxf(g2, g3));
  float e0 = __expf(g0 - mx), e1 = __expf(g1 - mx), e2 = __expf(g2 - mx), e3 = __expf(g3 - mx);
  float inv = 1.f / (e0 + e1 + e2 + e3);
  float g[4] = {e0 * inv, e1 * inv, e2 * inv, e3 * inv};

  int task = blockIdx.x * blockDim.x + threadIdx.x;
  const int total = NTN * KS_FPN * 32;
  if (task < total) {
    int lane = task & 31;
    int t = task >> 5;
    int ks = t % KS_FPN;
    int tn = t / KS_FPN;
    int n = tn * 16 + (lane & 15);
    int half = lane >> 4;
    int mat = ks / KS_D, ksl = ks % KS_D;
    const float* W = fpnW + (size_t)mat * D_ * D_;
    float sc = g[mat];
    __bf16* o = Bp + (size_t)task * 16;
#pragma unroll
    for (int p = 0; p < 8; ++p) {
      int k = ksl * 32 + half * 8 + ((p < 4) ? 2 * p : 2 * p + 8);
      o[2 * p]     = (__bf16)(W[(size_t)k * D_ + n] * sc);
      o[2 * p + 1] = (__bf16)(W[(size_t)(k + 1) * D_ + n] * sc);
    }
  } else if (task < total + D_) {
    int c = task - total;
    cbias[c] = g[0] * fpnB[c] + g[1] * fpnB[D_ + c] +
               g[2] * fpnB[2 * D_ + c] + g[3] * fpnB[3 * D_ + c];
  }
}

// ---------------------------------------------------------------------------
// Prep 0b: generic K=384 weight packer (row-major K x 384 fp32 -> bf16 frags)
// ---------------------------------------------------------------------------
__global__ void pack_b_kernel(const float* __restrict__ W, __bf16* __restrict__ dst) {
  int task = blockIdx.x * blockDim.x + threadIdx.x;
  const int total = NTN * KS_D * 32;
  if (task >= total) return;
  int lane = task & 31;
  int t = task >> 5;
  int ks = t % KS_D;
  int tn = t / KS_D;
  int n = tn * 16 + (lane & 15);
  int half = lane >> 4;
  __bf16* o = dst + (size_t)task * 16;
#pragma unroll
  for (int p = 0; p < 8; ++p) {
    int k = ks * 32 + half * 8 + ((p < 4) ? 2 * p : 2 * p + 8);
    o[2 * p]     = (__bf16)W[(size_t)k * D_ + n];
    o[2 * p + 1] = (__bf16)W[(size_t)(k + 1) * D_ + n];
  }
}

// ---------------------------------------------------------------------------
// Prep 0c: conv weight packer: (O,I,3) fp32 -> im2col B frags, K = 3*384
// im2col k -> (dp = k/384, ci = k%384); W[(o*384+ci)*3 + dp]
// ---------------------------------------------------------------------------
__global__ void pack_conv_b_kernel(const float* __restrict__ W, __bf16* __restrict__ dst) {
  int task = blockIdx.x * blockDim.x + threadIdx.x;
  const int total = NTN * KS_CONV * 32;
  if (task >= total) return;
  int lane = task & 31;
  int t = task >> 5;
  int ks = t % KS_CONV;
  int tn = t / KS_CONV;
  int o_ = tn * 16 + (lane & 15);
  int half = lane >> 4;
  __bf16* o = dst + (size_t)task * 16;
#pragma unroll
  for (int p = 0; p < 8; ++p) {
#pragma unroll
    for (int e = 0; e < 2; ++e) {
      int k = ks * 32 + half * 8 + ((p < 4) ? 2 * p : 2 * p + 8) + e;
      int dp = k / D_;
      int ci = k - dp * D_;
      o[2 * p + e] = (__bf16)W[((size_t)o_ * D_ + ci) * 3 + dp];
    }
  }
}

// ---------------------------------------------------------------------------
// Kernel 1: gated FPN GEMM, 16x64 strip per wave (A reused across 4 WMMAs)
// ---------------------------------------------------------------------------
__global__ void fpn_kernel(const float* __restrict__ h0, const float* __restrict__ h1,
                           const float* __restrict__ h2, const float* __restrict__ h3,
                           const __bf16* __restrict__ Bp, const float* __restrict__ cbias,
                           float* __restrict__ feat) {
  const float* hs[4] = {h0, h1, h2, h3};
  int wave = blockIdx.x * (blockDim.x >> 5) + (threadIdx.x >> 5);
  int tm = wave / 6, tg = wave % 6, tn0 = tg * 4;   // 1024 x 6 strips
  int lane = threadIdx.x & 31;

  v8f acc[4];
#pragma unroll
  for (int j = 0; j < 4; ++j) {
    float cb = cbias[(tn0 + j) * 16 + (lane & 15)];
#pragma unroll
    for (int v = 0; v < 8; ++v) acc[j][v] = cb;
  }

#pragma unroll
  for (int n = 0; n < 4; ++n) {
    const float* __restrict__ Hn = hs[n];
    for (int k0 = 0; k0 < D_; k0 += 32) {
      int ks = n * KS_D + (k0 >> 5);
      v16bf a = load_a_f32(Hn, D_, tm * 16, k0);
#pragma unroll
      for (int j = 0; j < 4; ++j) {
        v16bf b = load_b_packed(Bp, (tn0 + j) * KS_FPN + ks);
        acc[j] = wmma_bf16(a, b, acc[j]);
      }
    }
  }
#pragma unroll
  for (int j = 0; j < 4; ++j)
    store_d(feat, D_, tm * 16, (tn0 + j) * 16, acc[j], false);
}

// ---------------------------------------------------------------------------
// Kernel 2: SPN fused  x=relu(feat@Wsh+b) in LDS -> obj/reg dots -> decode
// 8 waves x 3 N-tiles; A fragment reused across the 3 tiles inside K-loop.
// ---------------------------------------------------------------------------
__global__ void spn_kernel(const float* __restrict__ feat,
                           const __bf16* __restrict__ Wshp, const float* __restrict__ bsh,
                           const float* __restrict__ Wobj, const float* __restrict__ bobj,
                           const float* __restrict__ Wreg, const float* __restrict__ breg,
                           const int* __restrict__ amask,
                           float* __restrict__ scr, float* __restrict__ props) {
  __shared__ float xs[16 * D_];
  __shared__ float objs[16 * A_];
  __shared__ float regs[16 * 2 * A_];

  int r0 = blockIdx.x * 16;
  int wv = threadIdx.x >> 5, lane = threadIdx.x & 31;

  v8f acc[3];
#pragma unroll
  for (int t = 0; t < 3; ++t) {
    float bb = bsh[(wv * 3 + t) * 16 + (lane & 15)];
#pragma unroll
    for (int v = 0; v < 8; ++v) acc[t][v] = bb;
  }
  for (int k0 = 0; k0 < D_; k0 += 32) {
    int ks = k0 >> 5;
    v16bf a = load_a_f32(feat, D_, r0, k0);
#pragma unroll
    for (int t = 0; t < 3; ++t) {
      v16bf b = load_b_packed(Wshp, (wv * 3 + t) * KS_D + ks);
      acc[t] = wmma_bf16(a, b, acc[t]);
    }
  }
#pragma unroll
  for (int t = 0; t < 3; ++t) {
    int n = (wv * 3 + t) * 16 + (lane & 15), mb = (lane >> 4) * 8;
#pragma unroll
    for (int v = 0; v < 8; ++v) xs[(mb + v) * D_ + n] = fmaxf(acc[t][v], 0.f);
  }
  __syncthreads();

  for (int task = threadIdx.x; task < 16 * (3 * A_); task += blockDim.x) {
    int row = task / (3 * A_);
    int o = task % (3 * A_);
    const float* xr = &xs[row * D_];
    float s = 0.f;
    if (o < A_) {
#pragma unroll 8
      for (int k = 0; k < D_; ++k) s += xr[k] * Wobj[k * A_ + o];
      objs[row * A_ + o] = s + bobj[o];
    } else {
      int j = o - A_;
#pragma unroll 8
      for (int k = 0; k < D_; ++k) s += xr[k] * Wreg[k * (2 * A_) + j];
      regs[row * 2 * A_ + j] = s + breg[j];
    }
  }
  __syncthreads();

  const float Ls[A_] = {1.f, 2.f, 3.f, 4.f, 6.f, 8.f, 12.f};
  for (int task = threadIdx.x; task < 16 * A_; task += blockDim.x) {
    int row = task / A_, a = task % A_;
    int gr = r0 + row;
    int b = gr / T_, tt = gr % T_;
    float obj = objs[row * A_ + a];
    if (amask[gr] == 0) obj = -1.0e9f;
    float rr0 = regs[row * 2 * A_ + 2 * a];
    float rr1 = regs[row * 2 * A_ + 2 * a + 1];
    float l = Ls[a];
    float c = (float)tt + 0.5f * (l - 1.f);
    float c2 = c + rr0 * l;
    float l2 = l * __expf(rr1);
    float ps = fminf(fmaxf(rintf(c2 - 0.5f * l2), 0.f), (float)(T_ - 1));
    float pe = fminf(fmaxf(rintf(c2 + 0.5f * l2), 0.f), (float)(T_ - 1));
    bool valid = (pe >= ps);
    int ai = b * NCAND + tt * A_ + a;
    scr[ai] = valid ? obj : -INFINITY;
    props[(size_t)ai * 2 + 0] = ps;
    props[(size_t)ai * 2 + 1] = pe;
  }
}

// ---------------------------------------------------------------------------
// Kernel 3: per-batch top-64 (iterative LDS argmax) + sequential NMS, keep 16
// ---------------------------------------------------------------------------
__global__ void topk_nms_kernel(const float* __restrict__ scr,
                                const float* __restrict__ props,
                                float* __restrict__ out, int* __restrict__ se) {
  __shared__ float s[NCAND];
  __shared__ float rv[256];
  __shared__ int ri[256];
  __shared__ float tval[TOPK_], tps[TOPK_], tpe[TOPK_];

  int b = blockIdx.x, tid = threadIdx.x;
  for (int j = tid; j < NCAND; j += blockDim.x) s[j] = scr[b * NCAND + j];
  __syncthreads();

  for (int it = 0; it < TOPK_; ++it) {
    float best = -INFINITY;
    int bi = -1;
    for (int j = tid; j < NCAND; j += blockDim.x) {
      float v = s[j];
      if (v > best) { best = v; bi = j; }
    }
    rv[tid] = best; ri[tid] = bi;
    __syncthreads();
    for (int off = 128; off > 0; off >>= 1) {
      if (tid < off) {
        if (rv[tid + off] > rv[tid]) { rv[tid] = rv[tid + off]; ri[tid] = ri[tid + off]; }
      }
      __syncthreads();
    }
    if (tid == 0) {
      int sel = ri[0];
      tval[it] = rv[0];
      if (sel >= 0) {
        tps[it] = props[(size_t)(b * NCAND + sel) * 2 + 0];
        tpe[it] = props[(size_t)(b * NCAND + sel) * 2 + 1];
        s[sel] = -INFINITY;
      } else {
        tps[it] = 0.f; tpe[it] = 0.f;
      }
    }
    __syncthreads();
  }

  if (tid == 0) {
    bool sup[TOPK_];
    for (int i = 0; i < TOPK_; ++i) sup[i] = false;
    for (int i = 0; i < TOPK_; ++i) {
      if (sup[i]) continue;
      for (int j = i + 1; j < TOPK_; ++j) {
        float inter = fmaxf(fminf(tpe[i], tpe[j]) - fmaxf(tps[i], tps[j]) + 1.f, 0.f);
        float li = fmaxf(tpe[i] - tps[i] + 1.f, 0.f);
        float lj = fmaxf(tpe[j] - tps[j] + 1.f, 0.f);
        float iou = inter / (li + lj - inter + 1e-6f);
        if (iou > 0.6f) sup[j] = true;
      }
    }
    int numk = 0;
    for (int i = 0; i < TOPK_; ++i) if (!sup[i]) numk++;
    int kept[KEEP_];
    int c = 0;
    for (int i = 0; i < TOPK_ && c < KEEP_; ++i) if (!sup[i]) kept[c++] = i;
    for (int i = 0; i < TOPK_ && c < KEEP_; ++i) if (sup[i]) kept[c++] = i;
    for (int k = 0; k < KEEP_; ++k) {
      bool valid = (k < numk);
      int id = kept[k];
      float ps = valid ? tps[id] : 0.f;
      float pe = valid ? tpe[id] : 0.f;
      float sc = valid ? tval[id] : 0.f;
      out[OFF_PROP + (b * KEEP_ + k) * 2 + 0] = ps;
      out[OFF_PROP + (b * KEEP_ + k) * 2 + 1] = pe;
      out[OFF_SCORES + b * KEEP_ + k] = sc;
      int si = (int)fminf(ps, pe), ei = (int)fmaxf(ps, pe);
      se[(b * KEEP_ + k) * 2 + 0] = si;
      se[(b * KEEP_ + k) * 2 + 1] = ei;
    }
  }
}

// ---------------------------------------------------------------------------
// Kernel 4: ROI align (16 sample points, linear interp)
// ---------------------------------------------------------------------------
__global__ void roi_kernel(const float* __restrict__ feat, const int* __restrict__ se,
                           float* __restrict__ roi) {
  int bk = blockIdx.x;
  int b = bk >> 4;
  int s0 = se[bk * 2], e0 = se[bk * 2 + 1];
  int seg = e0 - s0 + 1;
  for (int idx = threadIdx.x; idx < P_ * D_; idx += blockDim.x) {
    int p = idx / D_, d = idx % D_;
    float pos = ((float)p / 15.f) * (float)(seg - 1);
    int i0 = min((int)floorf(pos), seg - 1);
    int i1 = min(i0 + 1, seg - 1);
    float w = pos - (float)i0;
    float f0 = feat[(size_t)(b * T_ + s0 + i0) * D_ + d];
    float f1 = feat[(size_t)(b * T_ + s0 + i1) * D_ + d];
    roi[(size_t)(bk * P_ + p) * D_ + d] = (1.f - w) * f0 + w * f1;
  }
}

// ---------------------------------------------------------------------------
// Kernel 5: xb = mean_P(roi)
// ---------------------------------------------------------------------------
__global__ void xbmean_kernel(const float* __restrict__ roi, float* __restrict__ xb) {
  int idx = blockIdx.x * blockDim.x + threadIdx.x;
  if (idx >= B_ * KEEP_ * D_) return;
  int row = idx / D_, d = idx % D_;
  float sum = 0.f;
#pragma unroll
  for (int p = 0; p < P_; ++p) sum += roi[(size_t)(row * P_ + p) * D_ + d];
  xb[idx] = sum * (1.f / (float)P_);
}

// ---------------------------------------------------------------------------
// Kernel 6: M x 384 x 384 GEMM (packed B), 16x64 strip per wave, bias+ReLU
// ---------------------------------------------------------------------------
__global__ void gemm_bias_act_kernel(const float* __restrict__ Asrc,
                                     const __bf16* __restrict__ Wp,
                                     const float* __restrict__ bias,
                                     float* __restrict__ C, int M, int relu) {
  int wave = blockIdx.x * (blockDim.x >> 5) + (threadIdx.x >> 5);
  int tm = wave / 6, tg = wave % 6, tn0 = tg * 4;
  if (tm * 16 >= M) return;
  int lane = threadIdx.x & 31;
  v8f acc[4];
#pragma unroll
  for (int j = 0; j < 4; ++j) {
    float bb = bias[(tn0 + j) * 16 + (lane & 15)];
#pragma unroll
    for (int v = 0; v < 8; ++v) acc[j][v] = bb;
  }
  for (int k0 = 0; k0 < D_; k0 += 32) {
    int ks = k0 >> 5;
    v16bf a = load_a_f32(Asrc, D_, tm * 16, k0);
#pragma unroll
    for (int j = 0; j < 4; ++j) {
      v16bf b = load_b_packed(Wp, (tn0 + j) * KS_D + ks);
      acc[j] = wmma_bf16(a, b, acc[j]);
    }
  }
#pragma unroll
  for (int j = 0; j < 4; ++j)
    store_d(C, D_, tm * 16, (tn0 + j) * 16, acc[j], relu != 0);
}

// ---------------------------------------------------------------------------
// Kernel 7: cls (8) + bbox (2) heads, plain dots
// ---------------------------------------------------------------------------
__global__ void heads_small_kernel(const float* __restrict__ h2,
                                   const float* __restrict__ Wc, const float* __restrict__ bc,
                                   const float* __restrict__ Wr, const float* __restrict__ br,
                                   float* __restrict__ out) {
  int task = blockIdx.x * blockDim.x + threadIdx.x;
  if (task >= B_ * KEEP_ * 10) return;
  int row = task / 10, o = task % 10;
  const float* xr = h2 + (size_t)row * D_;
  float s = 0.f;
  if (o < 8) {
#pragma unroll 8
    for (int k = 0; k < D_; ++k) s += xr[k] * Wc[k * 8 + o];
    out[OFF_CLS + row * 8 + o] = s + bc[o];
  } else {
    int j = o - 8;
#pragma unroll 8
    for (int k = 0; k < D_; ++k) s += xr[k] * Wr[k * 2 + j];
    out[OFF_BOX + row * 2 + j] = s + br[j];
  }
}

// ---------------------------------------------------------------------------
// Kernel 8: 3-tap conv as im2col WMMA GEMM; A gathered w/ SAME padding,
// B prepacked; 16x64 strip per wave; fused ReLU.
// ---------------------------------------------------------------------------
__device__ __forceinline__ v16bf load_a_conv(const float* __restrict__ src, int bk, int k0) {
  int lane = threadIdx.x & 31;
  int p = lane & 15;
  int half = lane >> 4;
  v16bf a;
#pragma unroll
  for (int q = 0; q < 8; ++q) {
#pragma unroll
    for (int e = 0; e < 2; ++e) {
      int k = k0 + half * 8 + ((q < 4) ? 2 * q : 2 * q + 8) + e;
      int dp = k / D_;
      int ci = k - dp * D_;
      int ip = p + dp - 1;
      float v = (ip >= 0 && ip < P_) ? src[(size_t)(bk * P_ + ip) * D_ + ci] : 0.f;
      a[2 * q + e] = (__bf16)v;
    }
  }
  return a;
}

__global__ void conv3_kernel(const float* __restrict__ src, const __bf16* __restrict__ Wp,
                             const float* __restrict__ bias, float* __restrict__ dst) {
  int wave = blockIdx.x * (blockDim.x >> 5) + (threadIdx.x >> 5);
  int bk = wave / 6, tg = wave % 6, tn0 = tg * 4;   // bk in [0, 512)
  int lane = threadIdx.x & 31;
  v8f acc[4];
#pragma unroll
  for (int j = 0; j < 4; ++j) {
    float bb = bias[(tn0 + j) * 16 + (lane & 15)];
#pragma unroll
    for (int v = 0; v < 8; ++v) acc[j][v] = bb;
  }
  for (int k0 = 0; k0 < 3 * D_; k0 += 32) {
    int ks = k0 >> 5;
    v16bf a = load_a_conv(src, bk, k0);
#pragma unroll
    for (int j = 0; j < 4; ++j) {
      v16bf b = load_b_packed(Wp, (tn0 + j) * KS_CONV + ks);
      acc[j] = wmma_bf16(a, b, acc[j]);
    }
  }
#pragma unroll
  for (int j = 0; j < 4; ++j)
    store_d(dst, D_, bk * 16, (tn0 + j) * 16, acc[j], true);
}

// ---------------------------------------------------------------------------
// Kernel 9: mask output (1-channel 1x1 conv + sigmoid)
// ---------------------------------------------------------------------------
__global__ void maskout_kernel(const float* __restrict__ y2, const float* __restrict__ Wout,
                               const float* __restrict__ bout, float* __restrict__ out) {
  int idx = blockIdx.x * blockDim.x + threadIdx.x;
  if (idx >= B_ * KEEP_ * P_) return;
  const float* r = y2 + (size_t)idx * D_;
  float sum = bout[0];
#pragma unroll 8
  for (int k = 0; k < D_; ++k) sum += r[k] * Wout[k];
  out[OFF_MASK + idx] = 1.f / (1.f + __expf(-sum));
}

// ---------------------------------------------------------------------------
// Kernel 10: lengths = sum(attention_mask) per batch
// ---------------------------------------------------------------------------
__global__ void lengths_kernel(const int* __restrict__ am, float* __restrict__ out) {
  int b = threadIdx.x;
  if (b < B_) {
    int s = 0;
    for (int t = 0; t < T_; ++t) s += am[b * T_ + t];
    out[OFF_LEN + b] = (float)s;
  }
}

// ---------------------------------------------------------------------------
// Host launch
// ---------------------------------------------------------------------------
extern "C" void kernel_launch(void* const* d_in, const int* in_sizes, int n_in,
                              void* d_out, int out_size, void* d_ws, size_t ws_size,
                              hipStream_t stream) {
  (void)in_sizes; (void)n_in; (void)out_size; (void)ws_size;
  const float* hs6  = (const float*)d_in[0];
  const float* hs8  = (const float*)d_in[1];
  const float* hs10 = (const float*)d_in[2];
  const float* hs12 = (const float*)d_in[3];
  const int*   am   = (const int*)  d_in[4];
  const float* fpnW = (const float*)d_in[5];
  const float* fpnB = (const float*)d_in[6];
  const float* gate = (const float*)d_in[7];
  const float* shW  = (const float*)d_in[8];
  const float* shB  = (const float*)d_in[9];
  const float* objW = (const float*)d_in[10];
  const float* objB = (const float*)d_in[11];
  const float* regW = (const float*)d_in[12];
  const float* regB = (const float*)d_in[13];
  const float* fc1W = (const float*)d_in[14];
  const float* fc1B = (const float*)d_in[15];
  const float* fc2W = (const float*)d_in[16];
  const float* fc2B = (const float*)d_in[17];
  const float* clsW = (const float*)d_in[18];
  const float* clsB = (const float*)d_in[19];
  const float* bxW  = (const float*)d_in[20];
  const float* bxB  = (const float*)d_in[21];
  const float* c1W  = (const float*)d_in[22];
  const float* c1B  = (const float*)d_in[23];
  const float* c2W  = (const float*)d_in[24];
  const float* c2B  = (const float*)d_in[25];
  const float* moW  = (const float*)d_in[26];
  const float* moB  = (const float*)d_in[27];

  float* out = (float*)d_out;
  float* ws  = (float*)d_ws;

  size_t off = 0;
  float* feat  = ws + off; off += (size_t)B_ * T_ * D_;          // 6,291,456
  float* scr   = ws + off; off += (size_t)B_ * NCAND;
  float* props = ws + off; off += (size_t)B_ * NCAND * 2;
  int*   se    = (int*)(ws + off); off += (size_t)B_ * KEEP_ * 2;
  float* roi   = ws + off; off += (size_t)B_ * KEEP_ * P_ * D_;  // 3,145,728
  float* xb    = ws + off; off += (size_t)B_ * KEEP_ * D_;
  float* h1b   = ws + off; off += (size_t)B_ * KEEP_ * D_;
  float* h2b   = ws + off; off += (size_t)B_ * KEEP_ * D_;
  // packed bf16 weight regions (float-offset bookkeeping; counts are even)
  __bf16* fpnBp = (__bf16*)(ws + off); off += (size_t)NTN * KS_FPN  * 32 * 16 / 2;
  float*  cbias = ws + off;            off += D_;
  __bf16* shWp  = (__bf16*)(ws + off); off += (size_t)NTN * KS_D    * 32 * 16 / 2;
  __bf16* fc1p  = (__bf16*)(ws + off); off += (size_t)NTN * KS_D    * 32 * 16 / 2;
  __bf16* fc2p  = (__bf16*)(ws + off); off += (size_t)NTN * KS_D    * 32 * 16 / 2;
  __bf16* c1p   = (__bf16*)(ws + off); off += (size_t)NTN * KS_CONV * 32 * 16 / 2;
  __bf16* c2p   = (__bf16*)(ws + off); off += (size_t)NTN * KS_CONV * 32 * 16 / 2;
  // feat region is dead after ROI align -> reuse for the two mask-conv activations
  float* y1 = feat;
  float* y2 = feat + (size_t)B_ * KEEP_ * P_ * D_;

  // --- prep: pack all WMMA B operands to bf16 fragment layout ---
  {
    int t0 = NTN * KS_FPN * 32 + D_;
    fpn_pack_kernel<<<(t0 + 255) / 256, 256, 0, stream>>>(fpnW, fpnB, gate, fpnBp, cbias);
    int t1 = NTN * KS_D * 32;
    pack_b_kernel<<<(t1 + 255) / 256, 256, 0, stream>>>(shW,  shWp);
    pack_b_kernel<<<(t1 + 255) / 256, 256, 0, stream>>>(fc1W, fc1p);
    pack_b_kernel<<<(t1 + 255) / 256, 256, 0, stream>>>(fc2W, fc2p);
    int t2 = NTN * KS_CONV * 32;
    pack_conv_b_kernel<<<(t2 + 255) / 256, 256, 0, stream>>>(c1W, c1p);
    pack_conv_b_kernel<<<(t2 + 255) / 256, 256, 0, stream>>>(c2W, c2p);
  }

  // 1) gated FPN GEMM: 1024 M-tiles x 6 strips, 8 waves/block
  fpn_kernel<<<(1024 * 6) / 8, 256, 0, stream>>>(hs6, hs8, hs10, hs12, fpnBp, cbias, feat);
  // 2) SPN fused head + anchor decode
  spn_kernel<<<(B_ * T_) / 16, 256, 0, stream>>>(feat, shWp, shB, objW, objB, regW, regB,
                                                 am, scr, props);
  // 3) top-64 + NMS per batch
  topk_nms_kernel<<<B_, 256, 0, stream>>>(scr, props, out, se);
  // 4) ROI align
  roi_kernel<<<B_ * KEEP_, 256, 0, stream>>>(feat, se, roi);
  // 5) box head
  xbmean_kernel<<<((B_ * KEEP_ * D_) + 255) / 256, 256, 0, stream>>>(roi, xb);
  gemm_bias_act_kernel<<<((B_ * KEEP_ / 16) * 6) / 8, 256, 0, stream>>>(xb,  fc1p, fc1B, h1b, B_ * KEEP_, 1);
  gemm_bias_act_kernel<<<((B_ * KEEP_ / 16) * 6) / 8, 256, 0, stream>>>(h1b, fc2p, fc2B, h2b, B_ * KEEP_, 1);
  heads_small_kernel<<<((B_ * KEEP_ * 10) + 255) / 256, 256, 0, stream>>>(h2b, clsW, clsB, bxW, bxB, out);
  // 6) mask head: two im2col WMMA convs + sigmoid out
  conv3_kernel<<<((B_ * KEEP_) * 6) / 8, 256, 0, stream>>>(roi, c1p, c1B, y1);
  conv3_kernel<<<((B_ * KEEP_) * 6) / 8, 256, 0, stream>>>(y1,  c2p, c2B, y2);
  maskout_kernel<<<((B_ * KEEP_ * P_) + 255) / 256, 256, 0, stream>>>(y2, moW, moB, out);
  // 7) lengths
  lengths_kernel<<<1, 32, 0, stream>>>(am, out);
}
